// ContextRetentionLayer_78864189489922
// MI455X (gfx1250) — matrix operand order repaired and checked by reading, not compiled, not run on hardware
//
#include <hip/hip_runtime.h>
#include <stdint.h>

#define DIMD 1024   // embed dim
#define DIMW 512    // window
#define NROW 16384  // 4 * 4096 rows
#define MTILE 32    // rows per block (2 x 16-row WMMA M-tiles)

typedef __bf16 v16bf __attribute__((ext_vector_type(16)));
typedef float  v8f   __attribute__((ext_vector_type(8)));
typedef int    v4i   __attribute__((__vector_size__(16)));   // matches builtin proto

// ---- helpers -------------------------------------------------------------

__device__ __forceinline__ unsigned short f2bf(float f) {
  union { float f; uint32_t u; } c; c.f = f;
  uint32_t u = c.u;
  uint32_t r = u + 0x7FFFu + ((u >> 16) & 1u);   // round-to-nearest-even
  return (unsigned short)(r >> 16);
}

__device__ __forceinline__ float bf2f(unsigned short b) {
  union { uint32_t u; float f; } c;
  c.u = ((uint32_t)b) << 16;
  return c.f;
}

// A-matrix 16-bit 16x32 fragment (CDNA5 layout): lane half h holds
// K = 8h..8h+7 in VGPR0-3 and K = 16+8h..16+8h+7 in VGPR4-7.
__device__ __forceinline__ v16bf load_a_frag(const unsigned short* rowp, int k0, int half) {
  union { v16bf v; uint4 q[2]; } u;
  u.q[0] = *(const uint4*)(rowp + k0 + 8 * half);
  u.q[1] = *(const uint4*)(rowp + k0 + 16 + 8 * half);
  return u.v;
}

// B-matrix 16-bit 32x16 fragment: lane half h holds K = 8h..8h+7 and K = 16+8h..16+8h+7.
__device__ __forceinline__ v16bf load_b_frag(const unsigned short* colp, int k0, int half) {
  union { v16bf v; uint4 q[2]; } u;
  u.q[0] = *(const uint4*)(colp + k0 + 8 * half);
  u.q[1] = *(const uint4*)(colp + k0 + 16 + 8 * half);
  return u.v;
}

__device__ __forceinline__ v8f wmma_bf16(v16bf a, v16bf b, v8f c) {
  return __builtin_amdgcn_wmma_f32_16x16x32_bf16(false, a, false, b, (short)0, c, false, false);
}

// ---- gfx1250 async global->LDS copy (16B per lane) -----------------------

#if defined(__has_builtin)
#if __has_builtin(__builtin_amdgcn_global_load_async_to_lds_b128)
#define HAVE_ASYNC_B128 1
#endif
#if __has_builtin(__builtin_amdgcn_s_wait_asynccnt)
#define HAVE_WAIT_ASYNC 1
#endif
#endif

__device__ __forceinline__ void async_g2l_b128(const float* gsrc, float* ldst) {
#ifdef HAVE_ASYNC_B128
  __builtin_amdgcn_global_load_async_to_lds_b128(
      (__attribute__((address_space(1))) v4i*)gsrc,
      (__attribute__((address_space(3))) v4i*)ldst, 0, 0);
#else
  unsigned int loff =
      (unsigned int)(uintptr_t)(__attribute__((address_space(3))) void*)ldst;
  asm volatile("global_load_async_to_lds_b128 %0, %1, off"
               :: "v"(loff), "v"(gsrc) : "memory");
#endif
}

__device__ __forceinline__ void wait_async0() {
#ifdef HAVE_WAIT_ASYNC
  __builtin_amdgcn_s_wait_asynccnt(0);
#else
  asm volatile("s_wait_asynccnt 0x0" ::: "memory");
#endif
}

// ---- prep: fp32 -> bf16 copies of the small resident matrices ------------

__global__ __launch_bounds__(256) void crl_prep_mb(const float* __restrict__ mb,
                                                   unsigned short* __restrict__ mb_bf,
                                                   unsigned short* __restrict__ mbT_bf) {
  int i = blockIdx.x * 256 + threadIdx.x;
  if (i >= DIMW * DIMD) return;
  int w = i / DIMD, d = i % DIMD;
  unsigned short b = f2bf(mb[i]);
  mb_bf[i] = b;                 // [W][D] row-major (pass 1 B)
  mbT_bf[d * DIMW + w] = b;     // [D][W] transposed (pass 2 B, contiguous K)
}

__global__ __launch_bounds__(256) void crl_prep_gw(const float* __restrict__ gw,
                                                   unsigned short* __restrict__ gw_bf) {
  int i = blockIdx.x * 256 + threadIdx.x;
  if (i < DIMD * DIMD) gw_bf[i] = f2bf(gw[i]);
}

// ---- fused main kernel: one 32-row tile per block (8 wave32s) ------------
// LDS (160KB): sh_x 64KB | sh_sc 64KB (x-stage fp32 -> scores f32 -> gate bf16)
//              | sh_attn 32KB

__global__ __launch_bounds__(256) void crl_main(
    const float* __restrict__ x,                // [NROW][D] fp32
    const float* __restrict__ gate_b,           // [D]
    const unsigned short* __restrict__ mb_bf,   // [W][D] bf16
    const unsigned short* __restrict__ mbT_bf,  // [D][W] bf16
    const unsigned short* __restrict__ gw_bf,   // [E=D][D] bf16
    float* __restrict__ out)                    // [NROW][D] fp32
{
  __shared__ unsigned short sh_x[MTILE * DIMD];     // 64KB, x tile bf16
  __shared__ float          sh_sc[MTILE * DIMW];    // 64KB, multi-purpose
  __shared__ unsigned short sh_attn[MTILE * DIMW];  // 32KB, attn bf16

  unsigned short* sh_gate = (unsigned short*)sh_sc; // 32x1024 bf16 == 64KB
  float*          sh_stg  = sh_sc;                  // 16x1024 fp32 landing zone

  const int tid   = threadIdx.x;
  const int wid   = tid >> 5;     // 0..7
  const int lane  = tid & 31;
  const int half  = lane >> 4;    // 0/1
  const int lan16 = lane & 15;
  const long rowbase = (long)blockIdx.x * MTILE;

  // ---- stage x tile to LDS bf16 via async copies, 16 rows at a time ----
  for (int h = 0; h < 2; ++h) {
    const float* xsrc = x + (rowbase + h * 16) * DIMD;
    for (int f = tid; f < 16 * DIMD / 4; f += 256)
      async_g2l_b128(xsrc + f * 4, sh_stg + f * 4);
    wait_async0();
    __syncthreads();
    for (int f = tid; f < 16 * DIMD / 4; f += 256) {
      float4 v = *(const float4*)(sh_stg + f * 4);
      ushort4 b4;
      b4.x = f2bf(v.x); b4.y = f2bf(v.y); b4.z = f2bf(v.z); b4.w = f2bf(v.w);
      *(ushort4*)(sh_x + h * 16 * DIMD + f * 4) = b4;
    }
    __syncthreads();
  }

  // ---- pass 1: scores[32,512] = Xtile . mb^T  (waves split N: 64 cols) ----
  v8f sacc[2][4] = {};
  const int w0 = wid * 64;
  for (int k0 = 0; k0 < DIMD; k0 += 32) {
    v16bf a0 = load_a_frag(sh_x + lan16 * DIMD, k0, half);
    v16bf a1 = load_a_frag(sh_x + (16 + lan16) * DIMD, k0, half);
#pragma unroll
    for (int t = 0; t < 4; ++t) {
      v16bf b = load_b_frag(mb_bf + (long)(w0 + t * 16 + lan16) * DIMD, k0, half);
      sacc[0][t] = wmma_bf16(a0, b, sacc[0][t]);   // B reused across 2 M-tiles
      sacc[1][t] = wmma_bf16(a1, b, sacc[1][t]);
    }
  }
  // D layout: vgpr r holds (M = r + 8*half, N = lane&15)
#pragma unroll
  for (int mt = 0; mt < 2; ++mt)
#pragma unroll
    for (int t = 0; t < 4; ++t)
#pragma unroll
      for (int r = 0; r < 8; ++r) {
        int m = mt * 16 + r + 8 * half;
        int c = w0 + t * 16 + lan16;
        sh_sc[m * DIMW + c] = sacc[mt][t][r] * 0.03125f;   // 1/sqrt(1024)
      }
  __syncthreads();

  // ---- softmax over W: 8 lanes per row, shuffle reductions ----
  {
    const int row = tid >> 3;   // 0..31
    const int sub = tid & 7;
    float m = -3.0e38f;
    for (int c = sub; c < DIMW; c += 8) m = fmaxf(m, sh_sc[row * DIMW + c]);
    m = fmaxf(m, __shfl_xor(m, 4, 8));
    m = fmaxf(m, __shfl_xor(m, 2, 8));
    m = fmaxf(m, __shfl_xor(m, 1, 8));
    float s = 0.f;
    for (int c = sub; c < DIMW; c += 8) {
      float e = __expf(sh_sc[row * DIMW + c] - m);
      s += e;
      sh_attn[row * DIMW + c] = f2bf(e);   // unnormalized
    }
    s += __shfl_xor(s, 4, 8);
    s += __shfl_xor(s, 2, 8);
    s += __shfl_xor(s, 1, 8);
    float inv = 1.0f / s;
    for (int c = sub; c < DIMW; c += 8)    // normalize in place (own elems)
      sh_attn[row * DIMW + c] = f2bf(bf2f(sh_attn[row * DIMW + c]) * inv);
  }
  __syncthreads();   // scores dead -> sh_sc region becomes the gate buffer

  // ---- pass G: gate logits[32,1024] = Xtile . gw^T, sigmoid -> LDS bf16 ----
  const int n0 = wid * 128;
  {
    v8f gacc[2][8] = {};
    for (int k0 = 0; k0 < DIMD; k0 += 32) {
      v16bf a0 = load_a_frag(sh_x + lan16 * DIMD, k0, half);
      v16bf a1 = load_a_frag(sh_x + (16 + lan16) * DIMD, k0, half);
      if (k0 + 32 < DIMD)
        __builtin_prefetch(gw_bf + (long)(n0 + lan16) * DIMD + k0 + 32, 0, 1);
#pragma unroll
      for (int t = 0; t < 8; ++t) {
        v16bf b = load_b_frag(gw_bf + (long)(n0 + t * 16 + lan16) * DIMD, k0, half);
        gacc[0][t] = wmma_bf16(a0, b, gacc[0][t]);
        gacc[1][t] = wmma_bf16(a1, b, gacc[1][t]);
      }
    }
#pragma unroll
    for (int t = 0; t < 8; ++t) {
      int c = n0 + t * 16 + lan16;
      float bias = gate_b[c];
#pragma unroll
      for (int mt = 0; mt < 2; ++mt)
#pragma unroll
        for (int r = 0; r < 8; ++r) {
          int m = mt * 16 + r + 8 * half;
          float g = 1.0f / (1.0f + __expf(-(gacc[mt][t][r] + bias)));
          sh_gate[m * DIMD + c] = f2bf(g);
        }
    }
  }

  // ---- pass 2: retrieved[32,1024] = attn . mb ----
  v8f racc[2][8] = {};
  for (int k0 = 0; k0 < DIMW; k0 += 32) {
    v16bf a0 = load_a_frag(sh_attn + lan16 * DIMW, k0, half);
    v16bf a1 = load_a_frag(sh_attn + (16 + lan16) * DIMW, k0, half);
#pragma unroll
    for (int t = 0; t < 8; ++t) {
      v16bf b = load_b_frag(mbT_bf + (long)(n0 + t * 16 + lan16) * DIMW, k0, half);
      racc[0][t] = wmma_bf16(a0, b, racc[0][t]);
      racc[1][t] = wmma_bf16(a1, b, racc[1][t]);
    }
  }
  __syncthreads();   // order LDS gate writes vs reads (alias-safe)

  // ---- epilogue: out = g*x + (1-g)*retrieved (x re-read at fp32) ----
#pragma unroll
  for (int t = 0; t < 8; ++t) {
    int c = n0 + t * 16 + lan16;
#pragma unroll
    for (int mt = 0; mt < 2; ++mt)
#pragma unroll
      for (int r = 0; r < 8; ++r) {
        int m = mt * 16 + r + 8 * half;
        long grow = rowbase + m;
        float xv = x[grow * DIMD + c];
        float g = bf2f(sh_gate[m * DIMD + c]);
        out[grow * DIMD + c] = g * xv + (1.0f - g) * racc[mt][t][r];
      }
  }
}

// ---- host side -----------------------------------------------------------

extern "C" void kernel_launch(void* const* d_in, const int* in_sizes, int n_in,
                              void* d_out, int out_size, void* d_ws, size_t ws_size,
                              hipStream_t stream) {
  const float* x      = (const float*)d_in[0];   // [4,4096,1024]
  const float* mb     = (const float*)d_in[1];   // [512,1024]
  const float* gate_w = (const float*)d_in[2];   // [1024,1024]
  const float* gate_b = (const float*)d_in[3];   // [1024]
  float* out = (float*)d_out;

  // workspace layout (bf16 copies): mb_bf 1MB | mbT_bf 1MB | gw_bf 2MB
  char* ws = (char*)d_ws;
  unsigned short* mb_bf  = (unsigned short*)(ws);
  unsigned short* mbT_bf = (unsigned short*)(ws + (size_t)DIMW * DIMD * 2);
  unsigned short* gw_bf  = (unsigned short*)(ws + (size_t)2 * DIMW * DIMD * 2);

  crl_prep_mb<<<(DIMW * DIMD + 255) / 256, 256, 0, stream>>>(mb, mb_bf, mbT_bf);
  crl_prep_gw<<<(DIMD * DIMD + 255) / 256, 256, 0, stream>>>(gate_w, gw_bf);
  crl_main<<<NROW / MTILE, 256, 0, stream>>>(x, gate_b, mb_bf, mbT_bf, gw_bf, out);
}
